// PEConv_72181220376644
// MI455X (gfx1250) — matrix operand order repaired and checked by reading, not compiled, not run on hardware
//
#include <hip/hip_runtime.h>

#define NUM_NODES 50000
#define NUM_EDGES 800000

typedef __attribute__((ext_vector_type(16))) __bf16 v16bf;
typedef __attribute__((ext_vector_type(8)))  float  v8f;

union Frag {
    uint4 q[2];
    v16bf v;
};

__device__ __forceinline__ unsigned short f2bf(float f) {
    unsigned int u = __float_as_uint(f);
    if ((u & 0x7fffffffu) > 0x7f800000u) return (unsigned short)0x7fc0u; // NaN
    return (unsigned short)((u + 0x7fffu + ((u >> 16) & 1u)) >> 16);     // RNE
}

// ---------------------------------------------------------------------------
// Pack a row-major f32 weight matrix W[K][128] into bf16 B-fragment layout:
// element index = ((kt*8 + nt)*32 + lane)*16 + j
//   k = kt*32 + (lane/16)*8 + (j%8) + (j/8)*16 , n = nt*16 + (lane%16)
// so each lane's 16 bf16 (32 bytes) for a (kt,nt) fragment are contiguous.
// ---------------------------------------------------------------------------
__global__ void pack_weights(const float* __restrict__ W,
                             unsigned short* __restrict__ Wp, int K) {
    int idx = blockIdx.x * blockDim.x + threadIdx.x;
    if (idx >= K * 128) return;
    int j    = idx & 15;
    int lane = (idx >> 4) & 31;
    int blk  = idx >> 9;        // kt*8 + nt
    int nt   = blk & 7;
    int kt   = blk >> 3;
    int k = kt * 32 + ((lane >> 4) << 3) + (j & 7) + ((j >> 3) << 4);
    int n = nt * 16 + (lane & 15);
    Wp[idx] = f2bf(W[k * 128 + n]);
}

#define SA1 200   // 192 + 8 pad (ushorts), 16B-aligned rows
#define SA2 264   // 256 + 8 pad
#define SH  136   // 128 + 8 pad

// ---------------------------------------------------------------------------
// Edge kernel: 64 edges per block, 8 waves. Wave w: rows m0=(w/2)*16,
// column half ch=w&1 (ntiles ch*4..ch*4+3).
// ---------------------------------------------------------------------------
__global__ __launch_bounds__(256) void edge_kernel(
    const float* __restrict__ x, const float* __restrict__ pe,
    const long long* __restrict__ ei,
    const unsigned short* __restrict__ W1p, const float* __restrict__ b1,
    const unsigned short* __restrict__ W2p, const float* __restrict__ b2,
    float* __restrict__ agg)
{
    __shared__ __align__(16) unsigned short A[64 * SA1];
    __shared__ __align__(16) unsigned short H[64 * SH];
    __shared__ int DST[64];

    const int t = threadIdx.x;
    const long long e0 = (long long)blockIdx.x * 64;

    if (t < 64) DST[t] = (int)ei[NUM_EDGES + e0 + t];

    { // gather [x[src] | pe] -> bf16 LDS. 4 threads/row, 48 cols each.
        int row = t >> 2, part = t & 3;
        long long e = e0 + row;
        long long s = ei[e];
        const float* xr = x + s * 128;
        const float* pr = pe + e * 64;
        #pragma unroll
        for (int j = 0; j < 12; ++j) {
            int col = part * 48 + j * 4;   // 128-boundary falls on a j-step
            float4 v = (col < 128) ? *(const float4*)(xr + col)
                                   : *(const float4*)(pr + (col - 128));
            unsigned short* d = &A[row * SA1 + col];
            d[0] = f2bf(v.x); d[1] = f2bf(v.y); d[2] = f2bf(v.z); d[3] = f2bf(v.w);
        }
    }
    __syncthreads();

    const int lane = t & 31;
    const int w    = t >> 5;
    const int m0   = (w >> 1) * 16;
    const int ch   = w & 1;
    const int ln   = lane & 15;
    const int hsel = lane >> 4;
    const int arow = m0 + ln;

    // ---------------- GEMM1: [64x192]@[192x128], K=192 -> 6 k-steps --------
    v8f acc[4] = {};
    #pragma unroll
    for (int kt = 0; kt < 6; ++kt) {
        Frag fa;
        int cb = kt * 32 + 8 * hsel;
        fa.q[0] = *(const uint4*)&A[arow * SA1 + cb];
        fa.q[1] = *(const uint4*)&A[arow * SA1 + cb + 16];
        #pragma unroll
        for (int i = 0; i < 4; ++i) {
            int ntg = ch * 4 + i;
            Frag fb;
            const uint4* pb = (const uint4*)(W1p + (((kt * 8 + ntg) * 32 + lane) << 4));
            fb.q[0] = pb[0]; fb.q[1] = pb[1];
            acc[i] = __builtin_amdgcn_wmma_f32_16x16x32_bf16(
                false, fa.v, false, fb.v, (short)0, acc[i], false, false);
        }
    }
    // bias + ReLU -> bf16 staging (D layout: M = r + 8*hsel, N = ln)
    #pragma unroll
    for (int i = 0; i < 4; ++i) {
        int n = (ch * 4 + i) * 16 + ln;
        float bias = b1[n];
        #pragma unroll
        for (int r = 0; r < 8; ++r) {
            float v = acc[i][r] + bias;
            H[(m0 + hsel * 8 + r) * SH + n] = f2bf(v > 0.f ? v : 0.f);
        }
    }
    __syncthreads();

    // ---------------- GEMM2: [64x128]@[128x128], K=128 -> 4 k-steps --------
    v8f acc2[4] = {};
    #pragma unroll
    for (int kt = 0; kt < 4; ++kt) {
        Frag fa;
        int cb = kt * 32 + 8 * hsel;
        fa.q[0] = *(const uint4*)&H[arow * SH + cb];
        fa.q[1] = *(const uint4*)&H[arow * SH + cb + 16];
        #pragma unroll
        for (int i = 0; i < 4; ++i) {
            int ntg = ch * 4 + i;
            Frag fb;
            const uint4* pb = (const uint4*)(W2p + (((kt * 8 + ntg) * 32 + lane) << 4));
            fb.q[0] = pb[0]; fb.q[1] = pb[1];
            acc2[i] = __builtin_amdgcn_wmma_f32_16x16x32_bf16(
                false, fa.v, false, fb.v, (short)0, acc2[i], false, false);
        }
    }
    // bias + scatter-add messages to aggregated[dst]
    #pragma unroll
    for (int i = 0; i < 4; ++i) {
        int n = (ch * 4 + i) * 16 + ln;
        float bias = b2[n];
        #pragma unroll
        for (int r = 0; r < 8; ++r) {
            int m = m0 + hsel * 8 + r;
            atomicAdd(&agg[(long long)DST[m] * 128 + n], acc2[i][r] + bias);
        }
    }
}

// ---------------------------------------------------------------------------
// Node kernel: 64 nodes per block; A = [x | aggregated] (K=256), then U2.
// ---------------------------------------------------------------------------
__global__ __launch_bounds__(256) void node_kernel(
    const float* __restrict__ x, const float* __restrict__ agg,
    const unsigned short* __restrict__ U1p, const float* __restrict__ c1,
    const unsigned short* __restrict__ U2p, const float* __restrict__ c2,
    float* __restrict__ out)
{
    __shared__ __align__(16) unsigned short A[64 * SA2];
    __shared__ __align__(16) unsigned short H[64 * SH];

    const int t = threadIdx.x;
    const long long n0 = (long long)blockIdx.x * 64;

    { // gather [x[node] | agg[node]] -> bf16 LDS. 4 threads/row, 64 cols each.
        int row = t >> 2, part = t & 3;
        long long node = n0 + row;
        long long cl = node < NUM_NODES ? node : (NUM_NODES - 1);
        const float* xr = x   + cl * 128;
        const float* ar = agg + cl * 128;
        #pragma unroll
        for (int j = 0; j < 16; ++j) {
            int col = part * 64 + j * 4;   // parts 0,1 -> x ; parts 2,3 -> agg
            float4 v = (col < 128) ? *(const float4*)(xr + col)
                                   : *(const float4*)(ar + (col - 128));
            unsigned short* d = &A[row * SA2 + col];
            d[0] = f2bf(v.x); d[1] = f2bf(v.y); d[2] = f2bf(v.z); d[3] = f2bf(v.w);
        }
    }
    __syncthreads();

    const int lane = t & 31;
    const int w    = t >> 5;
    const int m0   = (w >> 1) * 16;
    const int ch   = w & 1;
    const int ln   = lane & 15;
    const int hsel = lane >> 4;
    const int arow = m0 + ln;

    // ---------------- GEMM U1: K=256 -> 8 k-steps --------------------------
    v8f acc[4] = {};
    #pragma unroll
    for (int kt = 0; kt < 8; ++kt) {
        Frag fa;
        int cb = kt * 32 + 8 * hsel;
        fa.q[0] = *(const uint4*)&A[arow * SA2 + cb];
        fa.q[1] = *(const uint4*)&A[arow * SA2 + cb + 16];
        #pragma unroll
        for (int i = 0; i < 4; ++i) {
            int ntg = ch * 4 + i;
            Frag fb;
            const uint4* pb = (const uint4*)(U1p + (((kt * 8 + ntg) * 32 + lane) << 4));
            fb.q[0] = pb[0]; fb.q[1] = pb[1];
            acc[i] = __builtin_amdgcn_wmma_f32_16x16x32_bf16(
                false, fa.v, false, fb.v, (short)0, acc[i], false, false);
        }
    }
    #pragma unroll
    for (int i = 0; i < 4; ++i) {
        int n = (ch * 4 + i) * 16 + ln;
        float bias = c1[n];
        #pragma unroll
        for (int r = 0; r < 8; ++r) {
            float v = acc[i][r] + bias;
            H[(m0 + hsel * 8 + r) * SH + n] = f2bf(v > 0.f ? v : 0.f);
        }
    }
    __syncthreads();

    // ---------------- GEMM U2: K=128 -> 4 k-steps --------------------------
    v8f acc2[4] = {};
    #pragma unroll
    for (int kt = 0; kt < 4; ++kt) {
        Frag fa;
        int cb = kt * 32 + 8 * hsel;
        fa.q[0] = *(const uint4*)&H[arow * SH + cb];
        fa.q[1] = *(const uint4*)&H[arow * SH + cb + 16];
        #pragma unroll
        for (int i = 0; i < 4; ++i) {
            int ntg = ch * 4 + i;
            Frag fb;
            const uint4* pb = (const uint4*)(U2p + (((kt * 8 + ntg) * 32 + lane) << 4));
            fb.q[0] = pb[0]; fb.q[1] = pb[1];
            acc2[i] = __builtin_amdgcn_wmma_f32_16x16x32_bf16(
                false, fa.v, false, fb.v, (short)0, acc2[i], false, false);
        }
    }
    #pragma unroll
    for (int i = 0; i < 4; ++i) {
        int n = (ch * 4 + i) * 16 + ln;
        float bias = c2[n];
        #pragma unroll
        for (int r = 0; r < 8; ++r) {
            long long node = n0 + m0 + hsel * 8 + r;
            if (node < NUM_NODES)
                out[node * 128 + n] = acc2[i][r] + bias;
        }
    }
}

extern "C" void kernel_launch(void* const* d_in, const int* in_sizes, int n_in,
                              void* d_out, int out_size, void* d_ws, size_t ws_size,
                              hipStream_t stream) {
    const float* x  = (const float*)d_in[0];
    const float* pe = (const float*)d_in[1];
    const float* W1 = (const float*)d_in[2];
    const float* b1 = (const float*)d_in[3];
    const float* W2 = (const float*)d_in[4];
    const float* b2 = (const float*)d_in[5];
    const float* U1 = (const float*)d_in[6];
    const float* c1 = (const float*)d_in[7];
    const float* U2 = (const float*)d_in[8];
    const float* c2 = (const float*)d_in[9];
    const long long* ei = (const long long*)d_in[10];
    float* out = (float*)d_out;

    // Workspace: aggregated f32 [N,128], then packed bf16 weights.
    float* agg = (float*)d_ws;
    unsigned short* W1p = (unsigned short*)(agg + (size_t)NUM_NODES * 128);
    unsigned short* W2p = W1p + 192 * 128;
    unsigned short* U1p = W2p + 128 * 128;
    unsigned short* U2p = U1p + 256 * 128;

    hipMemsetAsync(agg, 0, (size_t)NUM_NODES * 128 * sizeof(float), stream);

    pack_weights<<<(192 * 128) / 256, 256, 0, stream>>>(W1, W1p, 192);
    pack_weights<<<(128 * 128) / 256, 256, 0, stream>>>(W2, W2p, 128);
    pack_weights<<<(256 * 128) / 256, 256, 0, stream>>>(U1, U1p, 256);
    pack_weights<<<(128 * 128) / 256, 256, 0, stream>>>(U2, U2p, 128);

    edge_kernel<<<NUM_EDGES / 64, 256, 0, stream>>>(x, pe, ei, W1p, b1, W2p, b2, agg);
    node_kernel<<<(NUM_NODES + 63) / 64, 256, 0, stream>>>(x, agg, U1p, c1, U2p, c2, out);
}
